// PatchedFalconAttention_58102317580431
// MI455X (gfx1250) — matrix operand order repaired and checked by reading, not compiled, not run on hardware
//
#include <hip/hip_runtime.h>
#include <hip/hip_bf16.h>

// ---------------- problem constants (match reference) ----------------
#define B_    8
#define Q_    512
#define HID_  4096
#define NH_   32
#define DH_   128
#define BS_   64
#define NBLK_ 32
#define NUMB_ 256
#define T_    (B_ * Q_)             // 4096 tokens
#define KVC_  (NBLK_ * BS_)         // 2048 kv slots per sequence
#define QKVW_ ((NH_ + 2) * DH_)     // 4352 fused qkv width
#define QKVSTRIDE_ QKVW_

typedef __bf16 bf16;
typedef __attribute__((ext_vector_type(16))) __bf16 v16bf;
typedef __attribute__((ext_vector_type(8)))  float  v8f;
typedef __attribute__((ext_vector_type(4)))  unsigned int u32x4;
typedef __attribute__((ext_vector_type(4)))  int  i32x4;
typedef __attribute__((ext_vector_type(8)))  int  i32x8;

// =====================================================================
// TDM: build a D# for a 2D bf16 tile load (global -> LDS), optionally
// inserting LDS padding every tile row (pad fields per CDNA5 D# spec).
// =====================================================================
__device__ __forceinline__ void tdm_load_tile_bf16(
    unsigned lds_addr, const bf16* gptr,
    unsigned tensor_d0, unsigned tensor_d1,
    unsigned long long stride0_elems,
    unsigned tile_d0, unsigned tile_d1,
    unsigned pad_interval_code, unsigned pad_amount_code, bool pad_en) {
  unsigned long long ga = (unsigned long long)gptr;
  u32x4 g0;
  g0[0] = 1u;                                            // count=1 (valid), user mode
  g0[1] = lds_addr;                                      // lds_addr [63:32]
  g0[2] = (unsigned)(ga & 0xFFFFFFFFu);                  // global_addr lo
  g0[3] = (unsigned)((ga >> 32) & 0x1FFFFFFu)            // global_addr [56:32]
        | (2u << 30);                                    // type = 2 ("image")
  i32x8 g1;
  unsigned w0 = (1u << 16);                              // data_size = 1 (2 bytes)
  if (pad_en) w0 |= (1u << 20)                           // pad_enable
             | (pad_interval_code << 22)                 // pad_interval
             | (pad_amount_code << 25);                  // pad_amount
  g1[0] = (int)w0;
  g1[1] = (int)((tensor_d0 & 0xFFFFu) << 16);            // abar=0 | tensor_dim0[15:0]
  g1[2] = (int)(((tensor_d0 >> 16) & 0xFFFFu) | ((tensor_d1 & 0xFFFFu) << 16));
  g1[3] = (int)(((tensor_d1 >> 16) & 0xFFFFu) | ((tile_d0 & 0xFFFFu) << 16));
  g1[4] = (int)(tile_d1 & 0xFFFFu);                      // tile_dim1 | tile_dim2=0
  g1[5] = (int)(unsigned)(stride0_elems & 0xFFFFFFFFull);// tensor_dim0_stride lo
  g1[6] = (int)(unsigned)((stride0_elems >> 32) & 0xFFFFull);
  g1[7] = 0;
  i32x4 z4 = {0, 0, 0, 0};
#if defined(__clang_major__) && __clang_major__ >= 23
  i32x8 z8 = {0, 0, 0, 0, 0, 0, 0, 0};
  __builtin_amdgcn_tensor_load_to_lds(g0, g1, z4, z4, z8, 0);
#else
  __builtin_amdgcn_tensor_load_to_lds(g0, g1, z4, z4, 0);
#endif
}

// Per-lane async copy of 16 bytes global -> LDS (ASYNCcnt-tracked).
__device__ __forceinline__ void async_b128(unsigned lds_addr, const void* gptr) {
  unsigned long long ga = (unsigned long long)gptr;
  asm volatile("global_load_async_to_lds_b128 %0, %1, off"
               :: "v"(lds_addr), "v"(ga) : "memory");
}
__device__ __forceinline__ void wait_async0() {
  asm volatile("s_wait_asynccnt 0" ::: "memory");
}

// =====================================================================
// fp32 -> bf16 pre-cast (one-time; halves GEMM global traffic)
// =====================================================================
__global__ __launch_bounds__(256)
void cast_f32_bf16(const float* __restrict__ src, bf16* __restrict__ dst, size_t n) {
  size_t i = (size_t)blockIdx.x * 1024 + threadIdx.x * 4;
  if (i + 3 < n) {
#pragma unroll
    for (int j = 0; j < 4; ++j) dst[i + j] = (bf16)src[i + j];
  } else {
    for (; i < n; ++i) dst[i] = (bf16)src[i];
  }
}

// =====================================================================
// GEMM (NT): C[M,N] f32 = A[M,K]bf16 * B[N,K]bf16^T via WMMA.
// A tile: per-lane global_load_async_to_lds_b128. B tile: TDM with LDS
// row padding (80B stride). 128x64 block tile, 8 waves, 32x32 per wave.
// =====================================================================
#define GTM 128
#define GTN 64
#define GTK 32
#define LDA_PAD 8   // 80-byte LDS row stride (TDM pad: 4 DWORDs per 16 DWORDs)

__global__ __launch_bounds__(256)
void gemm_nt_bf16(const bf16* __restrict__ A, const bf16* __restrict__ Bm,
                  float* __restrict__ C, int M, int N, int K) {
  __shared__ __align__(16) bf16 As[GTM][GTK + LDA_PAD];
  __shared__ __align__(16) bf16 Bs[GTN][GTK + LDA_PAD];

  const int tid  = threadIdx.x;
  const int wave = tid >> 5;
  const int lane = tid & 31;
  const int bm = blockIdx.y * GTM;
  const int bn = blockIdx.x * GTN;
  const int wm = (wave >> 1) * 32;  // 4 waves along M
  const int wn = (wave & 1) * 32;   // 2 waves along N

  v8f acc[2][2] = {};

  const unsigned as_base = (unsigned)(unsigned long long)&As[0][0];
  const unsigned bs_base = (unsigned)(unsigned long long)&Bs[0][0];

  const int ar = tid >> 1;            // A rows: 2 threads/row, 16 elems each
  const int ac = (tid & 1) * 16;
  const unsigned lds_a = as_base + (unsigned)((ar * (GTK + LDA_PAD) + ac) * 2);

  const int lm = lane & 15;
  const int lh = (lane & 16) ? 8  : 0;  // unused; A and B frags both use kh
  const int kh = (lane & 16) ? 16 : 0;  // B-frag K sub-offset

  for (int k0 = 0; k0 < K; k0 += GTK) {
    // ---- B tile via Tensor Data Mover (wave 0 only; TENSORcnt) ----
    if (wave == 0) {
      tdm_load_tile_bf16(bs_base, Bm + (size_t)bn * K + k0,
                         (unsigned)(K - k0), (unsigned)(N - bn),
                         (unsigned long long)K,
                         GTK, GTN,
                         /*interval 16 DW=64B*/ 3u, /*amount 4 DW=16B*/ 3u, true);
    }
    // ---- A tile via per-lane async b128 (ASYNCcnt): 2 x 16B per thread ----
    {
      const bf16* ga = A + (size_t)(bm + ar) * K + k0 + ac;
      async_b128(lds_a, ga);
      async_b128(lds_a + 16, ga + 8);
    }
    wait_async0();
    if (wave == 0) __builtin_amdgcn_s_wait_tensorcnt((short)0);
    __syncthreads();

    // ---- fragments: contiguous k = 16h + s for both A and B ----
    v16bf af[2], bfg[2];
#pragma unroll
    for (int fm = 0; fm < 2; ++fm) {
      const int m = wm + fm * 16 + lm;
#pragma unroll
      for (int i = 0; i < 8; ++i) {
        af[fm][i]     = As[m][i + kh];
        af[fm][i + 8] = As[m][i + 8 + kh];
      }
    }
#pragma unroll
    for (int fn = 0; fn < 2; ++fn) {
      const int n = wn + fn * 16 + lm;
#pragma unroll
      for (int i = 0; i < 16; ++i) bfg[fn][i] = Bs[n][i + kh];
    }

#pragma unroll
    for (int fm = 0; fm < 2; ++fm)
#pragma unroll
      for (int fn = 0; fn < 2; ++fn)
        acc[fm][fn] = __builtin_amdgcn_wmma_f32_16x16x32_bf16(
            false, af[fm], false, bfg[fn], (short)0, acc[fm][fn], false, false);

    __syncthreads();
  }

  // ---- store C ----
  const int rowoff = (lane & 16) ? 8 : 0;
#pragma unroll
  for (int fm = 0; fm < 2; ++fm)
#pragma unroll
    for (int fn = 0; fn < 2; ++fn)
#pragma unroll
      for (int i = 0; i < 8; ++i) {
        const int m = bm + wm + fm * 16 + rowoff + i;
        const int n = bn + wn + fn * 16 + lm;
        C[(size_t)m * N + n] = acc[fm][fn][i];
      }
}

// =====================================================================
// Gather paged KV cache into contiguous per-batch bf16 buffers.
// =====================================================================
__global__ __launch_bounds__(256)
void gather_kv(const float* __restrict__ pk, const float* __restrict__ pv,
               const int* __restrict__ boff,
               bf16* __restrict__ kc, bf16* __restrict__ vc) {
  const size_t idx = (size_t)blockIdx.x * blockDim.x + threadIdx.x;
  if (idx >= (size_t)B_ * KVC_ * DH_) return;
  const int d  = (int)(idx % DH_);
  const size_t r = idx / DH_;
  const int kv = (int)(r % KVC_);
  const int b  = (int)(r / KVC_);
  const int blk = boff[b * NBLK_ + (kv / BS_)];
  const int off = kv % BS_;
  const size_t src = ((size_t)blk * BS_ + off) * DH_ + d;
  kc[idx] = (bf16)pk[src];
  vc[idx] = (bf16)pv[src];
}

// =====================================================================
// RoPE (in place on fused fp32 QKV in workspace) + scatter fresh K/V
// (bf16) into the gathered cache. One 256-thread block per token.
// =====================================================================
__global__ __launch_bounds__(256)
void rope_update(float* __restrict__ qkv, const int* __restrict__ pos_ids,
                 bf16* __restrict__ kc, bf16* __restrict__ vc) {
  const int t   = blockIdx.x;
  const int pos = pos_ids[t];
  const int b   = t / Q_;
  float* base = qkv + (size_t)t * QKVSTRIDE_;

  const float fpos = (float)pos;
  for (int p = threadIdx.x; p < NH_ * 64; p += 256) {
    const int hh = p >> 6;
    const int i  = p & 63;
    const float inv = __expf(-(2.0f * i / (float)DH_) * __logf(10000.0f));
    const float ang = fpos * inv;
    const float cs = __cosf(ang), sn = __sinf(ang);
    float* qp = base + hh * DH_;
    const float x0 = qp[i], x1 = qp[i + 64];
    qp[i]      = x0 * cs - x1 * sn;
    qp[i + 64] = x1 * cs + x0 * sn;
  }
  float* kp = base + NH_ * DH_;
  float* vp = kp + DH_;
  for (int i = threadIdx.x; i < 64; i += 256) {
    const float inv = __expf(-(2.0f * i / (float)DH_) * __logf(10000.0f));
    const float ang = fpos * inv;
    const float cs = __cosf(ang), sn = __sinf(ang);
    const float x0 = kp[i], x1 = kp[i + 64];
    kp[i]      = x0 * cs - x1 * sn;
    kp[i + 64] = x1 * cs + x0 * sn;
  }
  __syncthreads();
  const size_t row = ((size_t)b * KVC_ + pos) * DH_;
  for (int d = threadIdx.x; d < DH_; d += 256) {
    kc[row + d] = (bf16)kp[d];
    vc[row + d] = (bf16)vp[d];
  }
}

// =====================================================================
// Flash-style attention: one wave per (batch, head, 16-row q tile).
// QK^T and PV via v_wmma_f32_16x16x32_bf16; online softmax with wave32
// shuffles; P transposed C-layout -> A-layout through LDS. K/V read as
// bf16 directly (no per-element cvt in the hot loop). Output bf16.
// =====================================================================
__global__ __launch_bounds__(256)
void attention_kernel(const float* __restrict__ qkv,   // [T][(NH+2)*DH], rope applied
                      const bf16* __restrict__ kc,     // [B][KVC][DH]
                      const bf16* __restrict__ vc,     // [B][KVC][DH]
                      const int* __restrict__ hist,    // [B]
                      bf16* __restrict__ out) {        // [T][NH*DH]
  __shared__ bf16 pst[8][16][32 + 4];   // per-wave P staging (C->A transpose)

  const int wave = threadIdx.x >> 5;
  const int lane = threadIdx.x & 31;
  const int gw = blockIdx.x * 8 + wave;
  const int qt = gw & 31;          // q tile 0..31
  const int h  = (gw >> 5) & 31;   // head
  const int b  = gw >> 10;         // batch

  const int n16    = lane & 15;
  const int lh     = (lane & 16) ? 8  : 0;
  const int khi    = (lane & 16) ? 16 : 0;
  const int rowoff = (lane & 16) ? 8  : 0;

  const float scale = 0.088388347648318440550f;  // 1/sqrt(128)

  // ---- Q fragments (scale folded), A layout ----
  v16bf qa[4];
  {
    const size_t trow = ((size_t)(b * Q_ + qt * 16 + n16)) * QKVSTRIDE_ + (size_t)h * DH_;
#pragma unroll
    for (int kk = 0; kk < 4; ++kk)
#pragma unroll
      for (int i = 0; i < 8; ++i) {
        qa[kk][i]     = (bf16)(qkv[trow + kk * 32 + i + khi] * scale);
        qa[kk][i + 8] = (bf16)(qkv[trow + kk * 32 + i + 8 + khi] * scale);
      }
  }

  v8f oacc[8] = {};
  float rmax[8], rsum[8];
#pragma unroll
  for (int i = 0; i < 8; ++i) { rmax[i] = -1e30f; rsum[i] = 0.0f; }

  const int hb   = hist[b];
  const int qmax = hb + qt * 16 + 15;
  int nch = (qmax + 32) / 32;
  if (nch > KVC_ / 32) nch = KVC_ / 32;

  const bf16* kbase = kc + (size_t)b * KVC_ * DH_;
  const bf16* vbase = vc + (size_t)b * KVC_ * DH_;

  for (int c = 0; c < nch; ++c) {
    // ---- S[16x32] = Q * K^T ----
    v8f sc[2] = {};
#pragma unroll
    for (int sub = 0; sub < 2; ++sub) {
      const int kv = c * 32 + sub * 16 + n16;
#pragma unroll
      for (int kk = 0; kk < 4; ++kk) {
        v16bf kb;
        const bf16* kr = kbase + (size_t)kv * DH_ + kk * 32 + khi;
#pragma unroll
        for (int i = 0; i < 16; ++i) kb[i] = kr[i];
        sc[sub] = __builtin_amdgcn_wmma_f32_16x16x32_bf16(
            false, qa[kk], false, kb, (short)0, sc[sub], false, false);
      }
    }

    // ---- mask + online softmax ----
#pragma unroll
    for (int i = 0; i < 8; ++i) {
      const int m    = i + rowoff;
      const int qpos = hb + qt * 16 + m;
      const int kv0  = c * 32 + n16;
      float s0 = (kv0 > qpos)      ? -1e30f : sc[0][i];
      float s1 = (kv0 + 16 > qpos) ? -1e30f : sc[1][i];

      float v = fmaxf(s0, s1);
#pragma unroll
      for (int off = 1; off < 16; off <<= 1)
        v = fmaxf(v, __shfl_xor(v, off, 32));
      const float nmax = fmaxf(rmax[i], v);
      const float f = __expf(rmax[i] - nmax);
      rmax[i] = nmax;

      const float p0 = __expf(s0 - nmax);
      const float p1 = __expf(s1 - nmax);
      float rowp = p0 + p1;
#pragma unroll
      for (int off = 1; off < 16; off <<= 1)
        rowp += __shfl_xor(rowp, off, 32);
      rsum[i] = rsum[i] * f + rowp;

#pragma unroll
      for (int nd = 0; nd < 8; ++nd) oacc[nd][i] *= f;

      pst[wave][m][n16]      = (bf16)p0;
      pst[wave][m][16 + n16] = (bf16)p1;
    }

    // ---- P in A layout (same-wave LDS ops stay in order) ----
    v16bf pa;
#pragma unroll
    for (int i = 0; i < 8; ++i) {
      pa[i]     = pst[wave][n16][i + khi];
      pa[i + 8] = pst[wave][n16][i + 8 + khi];
    }

    // ---- O += P (16x32) * V (32x128) ----
#pragma unroll
    for (int nd = 0; nd < 8; ++nd) {
      v16bf vb;
#pragma unroll
      for (int i = 0; i < 16; ++i)
        vb[i] = vbase[((size_t)(c * 32 + i + khi)) * DH_ + nd * 16 + n16];
      oacc[nd] = __builtin_amdgcn_wmma_f32_16x16x32_bf16(
          false, pa, false, vb, (short)0, oacc[nd], false, false);
    }
  }

  // ---- normalize, store bf16 [t][h*DH + d] ----
#pragma unroll
  for (int i = 0; i < 8; ++i) {
    const int m = i + rowoff;
    const float inv = 1.0f / rsum[i];
    const size_t orow = ((size_t)(b * Q_ + qt * 16 + m)) * (NH_ * DH_) + (size_t)h * DH_;
#pragma unroll
    for (int nd = 0; nd < 8; ++nd)
      out[orow + nd * 16 + n16] = (bf16)(oacc[nd][i] * inv);
  }
}

// =====================================================================
// Launcher
// =====================================================================
extern "C" void kernel_launch(void* const* d_in, const int* in_sizes, int n_in,
                              void* d_out, int out_size, void* d_ws, size_t ws_size,
                              hipStream_t stream) {
  const float* hidden   = (const float*)d_in[0];   // [1, T, HID]
  const float* qkv_w    = (const float*)d_in[1];   // [4352, 4096]
  const float* dense_w  = (const float*)d_in[2];   // [4096, 4096]
  const float* past_k   = (const float*)d_in[3];   // [256, 64, 1, 128]
  const float* past_v   = (const float*)d_in[4];
  const int*   hist     = (const int*)d_in[7];     // [B]
  const int*   boff     = (const int*)d_in[8];     // [B, NBLK]
  const int*   pos_ids  = (const int*)d_in[9];     // [T]
  float* outp = (float*)d_out;

  char* ws = (char*)d_ws;
  auto align256 = [](size_t x) { return (x + 255) & ~(size_t)255; };
  size_t off = 0;
  bf16*  hid_bf  = (bf16*)(ws + off); off += align256((size_t)T_ * HID_ * 2);          // 33.5 MB
  bf16*  qkvw_bf = (bf16*)(ws + off); off += align256((size_t)QKVW_ * HID_ * 2);       // 35.7 MB
  bf16*  dw_bf   = (bf16*)(ws + off); off += align256((size_t)HID_ * HID_ * 2);        // 33.5 MB
  float* qkv_ws  = (float*)(ws + off); off += align256((size_t)T_ * QKVW_ * 4);        // 71.3 MB
  bf16*  kc_ws   = (bf16*)(ws + off); off += align256((size_t)B_ * KVC_ * DH_ * 2);    // 4.2 MB
  bf16*  vc_ws   = (bf16*)(ws + off); off += align256((size_t)B_ * KVC_ * DH_ * 2);    // 4.2 MB
  bf16*  att_bf  = (bf16*)(ws + off); off += align256((size_t)T_ * NH_ * DH_ * 2);     // 33.5 MB
  (void)ws_size; (void)in_sizes; (void)n_in; (void)out_size;

  // 0) pre-cast fp32 -> bf16 (activations + both weights)
  {
    size_t n0 = (size_t)T_ * HID_;
    size_t n1 = (size_t)QKVW_ * HID_;
    size_t n2 = (size_t)HID_ * HID_;
    cast_f32_bf16<<<(unsigned)((n0 + 1023) / 1024), 256, 0, stream>>>(hidden, hid_bf, n0);
    cast_f32_bf16<<<(unsigned)((n1 + 1023) / 1024), 256, 0, stream>>>(qkv_w, qkvw_bf, n1);
    cast_f32_bf16<<<(unsigned)((n2 + 1023) / 1024), 256, 0, stream>>>(dense_w, dw_bf, n2);
  }

  // 1) fused QKV GEMM: [T,4352] = hid_bf[T,4096] @ qkvw_bf[4352,4096]^T
  {
    dim3 grid(QKVW_ / GTN, T_ / GTM);
    gemm_nt_bf16<<<grid, 256, 0, stream>>>(hid_bf, qkvw_bf, qkv_ws, T_, QKVW_, HID_);
  }

  // 2) gather paged KV into contiguous per-batch bf16 buffers
  {
    const size_t n = (size_t)B_ * KVC_ * DH_;
    gather_kv<<<(unsigned)((n + 255) / 256), 256, 0, stream>>>(past_k, past_v, boff, kc_ws, vc_ws);
  }

  // 3) RoPE Q,K in place + scatter fresh K/V (bf16) into gathered cache
  rope_update<<<T_, 256, 0, stream>>>(qkv_ws, pos_ids, kc_ws, vc_ws);

  // 4) attention: 8192 waves, 8 waves/block -> bf16 output
  attention_kernel<<<(B_ * NH_ * 32) / 8, 256, 0, stream>>>(qkv_ws, kc_ws, vc_ws, hist, att_bf);

  // 5) output projection: d_out[T,4096] = att_bf[T,4096] @ dw_bf[4096,4096]^T
  {
    dim3 grid(HID_ / GTN, T_ / GTM);
    gemm_nt_bf16<<<grid, 256, 0, stream>>>(att_bf, dw_bf, outp, T_, HID_, HID_);
  }
}
